// I2DAttention_20332375179282
// MI455X (gfx1250) — matrix operand order repaired and checked by reading, not compiled, not run on hardware
//
#include <hip/hip_runtime.h>

typedef __attribute__((ext_vector_type(16))) _Float16 v16h;
typedef __attribute__((ext_vector_type(8)))  float    v8f;

// Fragment helper: 16 f16 = 8 VGPRs, matching WMMA A/B operand shape.
union Frag {
  v16h v;
  _Float16 h[16];
  unsigned u[8];
};

#define I_N   16
#define T_N   64
#define Q_N   196
#define K_N   32
#define E_N   512
#define H_N   8
#define D_N   64
#define QPAD  208          // 196 padded to 13 * 16
#define SLD   33           // padded LDS row stride (floats) to dodge bank conflicts

#define INV_SQRT_E 0.04419417382415922f   // 1/sqrt(512)
#define NEG_BIG    -3.402823466e38f

// Async memory -> LDS copy of 16 bytes (gfx1250, tracked by ASYNCcnt).
// ldsAddr: LDS byte offset (low 32 bits of a generic __shared__ address).
__device__ __forceinline__ void async_copy_b128(unsigned ldsAddr, const void* gaddr) {
  asm volatile("global_load_async_to_lds_b128 %0, %1, off"
               :: "v"(ldsAddr), "v"(gaddr) : "memory");
}
__device__ __forceinline__ void async_wait_all() {
  asm volatile("s_wait_asynccnt 0x0" ::: "memory");
}

// ---------------------------------------------------------------------------
// Stage A: per-head projection  Y[row, h*64+n] = sum_d X[row, h*64+d] * W[n,d]
// One 16x16 output tile per wave; K-loop = 64 in two 32-chunks of f16 WMMA.
// wave -> (mt, h, nt):  nt = wave&3, h = (wave>>2)&7, mt = wave>>5
// ---------------------------------------------------------------------------
__global__ void proj_f16_wmma(const float* __restrict__ X,
                              const float* __restrict__ W,
                              _Float16* __restrict__ Y) {
  const int wave  = blockIdx.x * (blockDim.x >> 5) + (threadIdx.x >> 5);
  const int lane  = threadIdx.x & 31;
  const int nt    = wave & 3;
  const int h     = (wave >> 2) & 7;
  const int mt    = wave >> 5;
  const int rowA  = mt * 16 + (lane & 15);
  const int halfA = (lane < 16) ? 0 : 8;    // A: K-offset per half-wave (ISA 7.12.2)
  const int halfB = (lane < 16) ? 0 : 8;    // B: K-offset per half-wave
  const int eCol  = nt * 16 + (lane & 15);

  v8f acc = {};
#pragma unroll
  for (int kb = 0; kb < D_N; kb += 32) {
    Frag a, b;
    const float* xr = X + (size_t)rowA * E_N + h * D_N;
#pragma unroll
    for (int p = 0; p < 8; ++p) {
      const int kk = ((p < 4) ? 2 * p : 2 * p + 8) + halfA + kb;
      const float2 f = *(const float2*)&xr[kk];
      a.h[2 * p]     = (_Float16)f.x;
      a.h[2 * p + 1] = (_Float16)f.y;
    }
    const float* wr = W + (size_t)eCol * D_N + kb + halfB;  // B[k=d][n=e] = W[e][d]
#pragma unroll
    for (int j = 0; j < 16; ++j) b.h[j] = (_Float16)wr[(j < 8) ? j : j + 8];
    acc = __builtin_amdgcn_wmma_f32_16x16x32_f16(false, a.v, false, b.v,
                                                 (short)0, acc, false, false);
  }
  const int col = nt * 16 + (lane & 15);
#pragma unroll
  for (int r = 0; r < 8; ++r) {
    const int rowD = mt * 16 + r + ((lane < 16) ? 0 : 8);
    Y[(size_t)rowD * E_N + h * D_N + col] = (_Float16)acc[r];
  }
}

// ---------------------------------------------------------------------------
// Stage B: one workgroup (128 threads / 4 waves) per (i, t, h).
//   Q/K/V tiles staged to LDS via GLOBAL_LOAD_ASYNC_TO_LDS_B128 (ASYNCcnt).
//   S = Q(196x64) @ K^T(64x32) via f16 WMMA (13x2 tiles, K-loop 2)
//   mask + scale, row softmax -> attention, col softmax -> text_attention,
//   column-mean of probs -> pooled = pa @ V (32x64 matvec), f16 out.
// ---------------------------------------------------------------------------
__global__ void attn_core(const _Float16* __restrict__ qh,
                          const _Float16* __restrict__ kh,
                          const _Float16* __restrict__ vh,
                          const int* __restrict__ tmask,
                          float* __restrict__ attnOut,
                          float* __restrict__ txOut,
                          _Float16* __restrict__ pooledH) {
  __shared__ __align__(16) _Float16 Qs[QPAD * D_N];
  __shared__ __align__(16) _Float16 Ks[K_N * D_N];
  __shared__ __align__(16) _Float16 Vs[K_N * D_N];
  __shared__ float Sm[QPAD * SLD];   // masked+scaled energy
  __shared__ float Pm[QPAD * SLD];   // row-softmax probabilities
  __shared__ float colmax[K_N], colisum[K_N], pa[K_N];
  __shared__ int   maskS[K_N];

  const int tid = threadIdx.x;
  const int bid = blockIdx.x;
  const int h = bid & 7;
  const int t = (bid >> 3) & 63;
  const int i = bid >> 9;

  // ---- stage LDS tiles: async mem->LDS, 16B (8 f16) per chunk ----
  for (int c = tid; c < (QPAD * D_N) / 8; c += blockDim.x) {
    const int row = c >> 3;
    const int d0  = (c & 7) * 8;
    if (row < Q_N) {
      async_copy_b128((unsigned)(uintptr_t)&Qs[row * D_N + d0],
                      &qh[((size_t)(i * Q_N + row) * E_N) + h * D_N + d0]);
    } else {
      *(uint4*)&Qs[row * D_N + d0] = make_uint4(0u, 0u, 0u, 0u);  // zero pad rows
    }
  }
  for (int c = tid; c < (K_N * D_N) / 8; c += blockDim.x) {
    const int row = c >> 3;
    const int d0  = (c & 7) * 8;
    const size_t g = ((size_t)(t * K_N + row) * E_N) + h * D_N + d0;
    async_copy_b128((unsigned)(uintptr_t)&Ks[row * D_N + d0], &kh[g]);
    async_copy_b128((unsigned)(uintptr_t)&Vs[row * D_N + d0], &vh[g]);
  }
  if (tid < K_N) maskS[tid] = tmask[t * K_N + tid];
  async_wait_all();          // ASYNCcnt not covered by __syncthreads
  __syncthreads();

  // ---- energy: 26 tiles of 16x16, distributed across 4 waves ----
  const int lane  = tid & 31;
  const int wv    = tid >> 5;
  const int halfA = (lane < 16) ? 0 : 8;
  const int halfB = (lane < 16) ? 0 : 8;
  for (int tt = wv; tt < 26; tt += 4) {
    const int mt   = tt >> 1;
    const int nt   = tt & 1;
    const int rowA = mt * 16 + (lane & 15);
    const int keyC = nt * 16 + (lane & 15);
    v8f acc = {};
#pragma unroll
    for (int kb = 0; kb < D_N; kb += 32) {
      Frag a, b;
#pragma unroll
      for (int p = 0; p < 8; ++p) {
        const int kk = ((p < 4) ? 2 * p : 2 * p + 8) + halfA + kb;
        a.u[p] = *(const unsigned*)&Qs[rowA * D_N + kk];
      }
#pragma unroll
      for (int p = 0; p < 8; ++p) {          // B[k=d][n=key] = K[key][d]
        const int dd = kb + ((p < 4) ? 2 * p : 2 * p + 8) + halfB;
        b.u[p] = *(const unsigned*)&Ks[keyC * D_N + dd];
      }
      acc = __builtin_amdgcn_wmma_f32_16x16x32_f16(false, a.v, false, b.v,
                                                   (short)0, acc, false, false);
    }
    const bool ok = (maskS[keyC] != 0);
#pragma unroll
    for (int r = 0; r < 8; ++r) {
      const int rr = mt * 16 + r + ((lane < 16) ? 0 : 8);
      const float e = ok ? acc[r] : NEG_BIG;
      Sm[rr * SLD + keyC] = e * INV_SQRT_E;
    }
  }
  __syncthreads();

  // ---- row softmax over k -> attention ----
  const size_t aBase = ((((size_t)i * T_N + t) * H_N + h) * Q_N) * K_N;
  for (int r = tid; r < Q_N; r += blockDim.x) {
    float m = NEG_BIG;
#pragma unroll
    for (int k = 0; k < K_N; ++k) m = fmaxf(m, Sm[r * SLD + k]);
    float s = 0.f;
#pragma unroll
    for (int k = 0; k < K_N; ++k) {
      const float e = __expf(Sm[r * SLD + k] - m);
      Pm[r * SLD + k] = e;
      s += e;
    }
    const float inv = 1.0f / s;
#pragma unroll
    for (int k = 0; k < K_N; ++k) {
      const float p = Pm[r * SLD + k] * inv;
      Pm[r * SLD + k] = p;
      attnOut[aBase + (size_t)r * K_N + k] = p;
    }
  }
  __syncthreads();

  // ---- column stats over q (32 columns) + pooled attention ----
  if (tid < K_N) {
    const int c = tid;
    float m = NEG_BIG;
    for (int q = 0; q < Q_N; ++q) m = fmaxf(m, Sm[q * SLD + c]);
    float s = 0.f;
    for (int q = 0; q < Q_N; ++q) s += __expf(Sm[q * SLD + c] - m);
    colmax[c]  = m;
    colisum[c] = 1.0f / s;
    float ps = 0.f;
    for (int q = 0; q < Q_N; ++q) ps += Pm[q * SLD + c];
    pa[c] = ps * (1.0f / (float)Q_N);   // mean over q of attention probs
  }
  __syncthreads();

  // ---- text_attention (col softmax) ----
  for (int idx = tid; idx < Q_N * K_N; idx += blockDim.x) {
    const int q = idx >> 5, k = idx & 31;
    txOut[aBase + idx] = __expf(Sm[q * SLD + k] - colmax[k]) * colisum[k];
  }

  // ---- pooled = pa @ V  (mean_q(attn @ V) == (mean_q attn) @ V) ----
  if (tid < D_N) {
    float s = 0.f;
#pragma unroll
    for (int k = 0; k < K_N; ++k) s += pa[k] * (float)Vs[k * D_N + tid];
    pooledH[((size_t)(i * T_N + t)) * E_N + h * D_N + tid] = (_Float16)s;
  }
}

// ---------------------------------------------------------------------------
// Stage C: out(1024x512) = pooled(1024x512,f16) @ Wo^T(512x512,f32->f16) + bo
// One 16x16 tile per wave; K-loop = 512 in 32-chunks.
// wave -> (mt, nt): nt = wave&31, mt = wave>>5
// ---------------------------------------------------------------------------
__global__ void fc_out_wmma(const _Float16* __restrict__ P,
                            const float* __restrict__ Wo,
                            const float* __restrict__ bo,
                            float* __restrict__ out) {
  const int wave  = blockIdx.x * (blockDim.x >> 5) + (threadIdx.x >> 5);
  const int lane  = threadIdx.x & 31;
  const int nt    = wave & 31;
  const int mt    = wave >> 5;
  const int rowA  = mt * 16 + (lane & 15);
  const int fCol  = nt * 16 + (lane & 15);
  const int halfA = (lane < 16) ? 0 : 8;
  const int halfB = (lane < 16) ? 0 : 8;

  v8f acc = {};
  for (int kb = 0; kb < E_N; kb += 32) {
    Frag a, b;
#pragma unroll
    for (int p = 0; p < 8; ++p) {
      const int kk = ((p < 4) ? 2 * p : 2 * p + 8) + halfA + kb;
      a.u[p] = *(const unsigned*)&P[(size_t)rowA * E_N + kk];
    }
    const float* wr = Wo + (size_t)fCol * E_N + kb + halfB;  // B[k=e][n=f] = Wo[f][e]
#pragma unroll
    for (int j = 0; j < 16; ++j) b.h[j] = (_Float16)wr[(j < 8) ? j : j + 8];
    acc = __builtin_amdgcn_wmma_f32_16x16x32_f16(false, a.v, false, b.v,
                                                 (short)0, acc, false, false);
  }
  const float bias = bo[fCol];
#pragma unroll
  for (int r = 0; r < 8; ++r) {
    const int rowD = mt * 16 + r + ((lane < 16) ? 0 : 8);
    out[(size_t)rowD * E_N + fCol] = acc[r] + bias;
  }
}

// ---------------------------------------------------------------------------
extern "C" void kernel_launch(void* const* d_in, const int* in_sizes, int n_in,
                              void* d_out, int out_size, void* d_ws, size_t ws_size,
                              hipStream_t stream) {
  (void)in_sizes; (void)n_in; (void)out_size; (void)ws_size;

  const float* image = (const float*)d_in[0];   // (16,196,512)
  const float* text  = (const float*)d_in[1];   // (64,32,512)
  const int*   tmask = (const int*)d_in[2];     // (64,32) bool -> int
  const float* Wq    = (const float*)d_in[3];   // (64,64)
  const float* Wk    = (const float*)d_in[4];
  const float* Wv    = (const float*)d_in[5];
  const float* Wo    = (const float*)d_in[6];   // (512,512)
  const float* bo    = (const float*)d_in[7];   // (512,)

  float* out     = (float*)d_out;                                    // (16,64,512)
  float* attnOut = out + (size_t)I_N * T_N * E_N;                    // (16,64,8,196,32)
  float* txOut   = attnOut + (size_t)I_N * T_N * H_N * Q_N * K_N;    // same shape

  _Float16* qh = (_Float16*)d_ws;                          // 3136*512 f16
  _Float16* kh = qh + (size_t)I_N * Q_N * E_N;             // 2048*512
  _Float16* vh = kh + (size_t)T_N * K_N * E_N;             // 2048*512
  _Float16* ph = vh + (size_t)T_N * K_N * E_N;             // 1024*512

  // Stage A: projections (q: 196 Mtiles, k/v: 128 Mtiles; 8 waves/block)
  proj_f16_wmma<<<196 * 4, 256, 0, stream>>>(image, Wq, qh);
  proj_f16_wmma<<<128 * 4, 256, 0, stream>>>(text,  Wk, kh);
  proj_f16_wmma<<<128 * 4, 256, 0, stream>>>(text,  Wv, vh);

  // Stage B: 8192 workgroups = (i,t,h)
  attn_core<<<I_N * T_N * H_N, 128, 0, stream>>>(qh, kh, vh, tmask,
                                                 attnOut, txOut, ph);

  // Stage C: 2048 tiles, 8 waves/block
  fc_out_wmma<<<256, 256, 0, stream>>>(ph, Wo, bo, out);
}
